// DSSLayer_24618752540895
// MI455X (gfx1250) — matrix operand order repaired and checked by reading, not compiled, not run on hardware
//
#include <hip/hip_runtime.h>
#include <hip/hip_bf16.h>
#include <stdint.h>

// Problem constants (match reference)
#define B_  4
#define L_  4096
#define H_  1024
#define N_  64
#define LK_ 512

// Padded transposed-u layout: u_t[b][h][PADL + t], row length ROW halfs.
#define PADL 512
#define ROW  4640   // 512 + 4096 + 32 ; 9280 bytes, 16B-aligned rows

typedef _Float16 v8h  __attribute__((ext_vector_type(8)));
typedef _Float16 v16h __attribute__((ext_vector_type(16)));
typedef float    v8f  __attribute__((ext_vector_type(8)));

// ---------------------------------------------------------------------------
// Kernel 0: zero-fill the padded u_t buffer (pads must be 0; ws is poisoned).
// ---------------------------------------------------------------------------
__global__ void dss_zfill(uint32_t* __restrict__ p, size_t ndwords) {
    size_t i = (size_t)blockIdx.x * blockDim.x + threadIdx.x;
    size_t stride = (size_t)gridDim.x * blockDim.x;
    for (; i < ndwords; i += stride) p[i] = 0u;
}

// ---------------------------------------------------------------------------
// Kernel 1: transpose u (B,L,H) f32 -> u_t (B,H,PADL+L) f16 via LDS tiles.
// block (32,8), grid (H/32, L/32, B)
// ---------------------------------------------------------------------------
__global__ void dss_utrans(const float* __restrict__ u, _Float16* __restrict__ uT) {
    __shared__ float tile[32][33];
    const int b  = blockIdx.z;
    const int h0 = blockIdx.x * 32;
    const int t0 = blockIdx.y * 32;
    const int tx = threadIdx.x, tyb = threadIdx.y;
#pragma unroll
    for (int ph = 0; ph < 4; ++ph) {
        int ty = tyb + ph * 8;
        tile[ty][tx] = u[((size_t)(b * L_ + t0 + ty)) * H_ + (h0 + tx)]; // coalesced over h
    }
    __syncthreads();
#pragma unroll
    for (int ph = 0; ph < 4; ++ph) {
        int hy = tyb + ph * 8;
        uT[((size_t)(b * H_ + h0 + hy)) * ROW + (size_t)(PADL + t0 + tx)] =
            (_Float16)tile[tx][hy];                                      // coalesced over t
    }
}

// ---------------------------------------------------------------------------
// Kernel 2: DSS 'exp' kernel generation  k[h][l], l in [0,512), f16 output.
// ---------------------------------------------------------------------------
__global__ void dss_kgen(const float* __restrict__ Lre, const float* __restrict__ Lim,
                         const float* __restrict__ logdt,
                         const float* __restrict__ Wre, const float* __restrict__ Wim,
                         _Float16* __restrict__ kout) {
    const int h   = blockIdx.x;
    const int tid = threadIdx.x;
    __shared__ float sA[N_], sB[N_], sWr[N_], sWi[N_];
    const float dt = __expf(logdt[h]);
    if (tid < N_) {
        sA[tid]  = -__expf(Lre[tid]) * dt;   // Re(dt*Lambda) < 0
        sB[tid]  = Lim[tid] * dt;            // Im(dt*Lambda)
        sWr[tid] = Wre[h * N_ + tid];
        sWi[tid] = Wim[h * N_ + tid];
    }
    __syncthreads();
    for (int l = tid; l < LK_; l += 128) {
        const float lf = (float)l;
        float acc = 0.f;
#pragma unroll 8
        for (int n = 0; n < N_; ++n) {
            float ea = __expf(sA[n] * lf);
            float s, c;
            __sincosf(sB[n] * lf, &s, &c);
            acc += ea * (sWr[n] * c - sWi[n] * s);   // Re(W * e^{dtLam*l})
        }
        kout[(size_t)h * LK_ + l] = (_Float16)acc;
    }
}

// ---------------------------------------------------------------------------
// Kernel 3: causal FIR via WMMA.  One block per (h,b), 256 thr = 8 waves.
// Wave w owns output tiles q = w and q = w+8 (256 times each).
//   A_c[i,r] = k[i - r + 32c]          (LDS-baked per (h,c))
//   B_c[r,j] = srow[PADL + T - 32c + 16j + r]   (u_t row async-staged in LDS)
//   Y[i,j]  -> t = T + 16j + i
// c is the OUTER loop so only a few A fragments are live (no spills); the
// two per-wave accumulator chains interleave to hide WMMA latency.
// ---------------------------------------------------------------------------
__global__ void dss_conv(const _Float16* __restrict__ uT,
                         const _Float16* __restrict__ kf,
                         const float* __restrict__ u,
                         const float* __restrict__ Dv,
                         float* __restrict__ out) {
    const int h    = blockIdx.x;
    const int b    = blockIdx.y;
    const int tid  = threadIdx.x;
    const int wave = tid >> 5;
    const int lane = tid & 31;
    const int j    = lane & 15;   // output-block column (N)
    const int hi   = lane >> 4;   // lane-half selector

    __shared__ __align__(32) _Float16 skl[LK_];
    __shared__ __align__(32) _Float16 afrag[17][512];
    __shared__ __align__(32) _Float16 srow[ROW];

    // --- async-stage the whole u_t row for (b,h) into LDS: 580 x 16B ------
    {
        const _Float16* grow  = uT + ((size_t)(b * H_ + h)) * ROW;
        const uint32_t  lbase = (uint32_t)(uintptr_t)&srow[0]; // LDS byte offset
        for (int i = tid; i < ROW / 8; i += 256) {
            const _Float16* g = grow + (size_t)i * 8;
            uint32_t        l = lbase + (uint32_t)i * 16;
            asm volatile("global_load_async_to_lds_b128 %0, %1, off"
                         :: "v"(l), "v"(g) : "memory");
        }
        asm volatile("s_wait_asynccnt 0x0" ::: "memory");
    }

    // --- stage k[h] and bake A fragments ----------------------------------
    for (int i = tid; i < LK_; i += 256) skl[i] = kf[(size_t)h * LK_ + i];
    __syncthreads();

    for (int idx = tid; idx < 17 * 512; idx += 256) {
        const int c  = idx >> 9;
        const int w  = idx & 511;
        const int ln = w >> 4;
        const int e  = w & 15;
        const int hh = ln >> 4;
        const int M  = ln & 15;
        const int K  = (e < 8) ? (8 * hh + e) : (8 * hh + e + 8);
        const int kk = M - K + 32 * c;
        afrag[c][w] = (kk >= 0 && kk < LK_) ? skl[kk] : (_Float16)0.f;
    }
    __syncthreads();

    const float Dh   = Dv[h];
    const int   T0   = wave << 8;          // tile q = wave
    const int   T1   = (wave + 8) << 8;    // tile q = wave + 8
    const int   boff = 16 * j + 8 * hi;    // per-lane half offset into B run

    __builtin_prefetch(&u[((size_t)b * L_ + T0) * H_ + h], 0, 3);

    v8f acc0 = {}, acc1 = {};
#pragma unroll 4
    for (int c = 0; c < 17; ++c) {
        v16h a = *(const v16h*)(&afrag[c][lane * 16]);   // one A per c, 2 uses
        {
            const _Float16* p = &srow[PADL + T0 + boff - 32 * c];
            v8h x0 = *(const v8h*)p;
            v8h x1 = *(const v8h*)(p + 16);
            v16h bb = __builtin_shufflevector(x0, x1,
                        0, 1, 2, 3, 4, 5, 6, 7, 8, 9, 10, 11, 12, 13, 14, 15);
            acc0 = __builtin_amdgcn_wmma_f32_16x16x32_f16(
                       false, a, false, bb, (short)0, acc0, false, false);
        }
        {
            const _Float16* p = &srow[PADL + T1 + boff - 32 * c];
            v8h x0 = *(const v8h*)p;
            v8h x1 = *(const v8h*)(p + 16);
            v16h bb = __builtin_shufflevector(x0, x1,
                        0, 1, 2, 3, 4, 5, 6, 7, 8, 9, 10, 11, 12, 13, 14, 15);
            acc1 = __builtin_amdgcn_wmma_f32_16x16x32_f16(
                       false, a, false, bb, (short)0, acc1, false, false);
        }
    }

    // --- store: D rows M = 8*hi + v, N = j -> t = T + 16j + 8hi + v -------
    const int tb0 = T0 + 16 * j + 8 * hi;
    const int tb1 = T1 + 16 * j + 8 * hi;
#pragma unroll
    for (int v = 0; v < 8; ++v) {
        const size_t oi = ((size_t)b * L_ + (size_t)(tb0 + v)) * H_ + h;
        out[oi] = acc0[v] + Dh * u[oi];
    }
#pragma unroll
    for (int v = 0; v < 8; ++v) {
        const size_t oi = ((size_t)b * L_ + (size_t)(tb1 + v)) * H_ + h;
        out[oi] = acc1[v] + Dh * u[oi];
    }
}

// ---------------------------------------------------------------------------
extern "C" void kernel_launch(void* const* d_in, const int* in_sizes, int n_in,
                              void* d_out, int out_size, void* d_ws, size_t ws_size,
                              hipStream_t stream) {
    const float* u     = (const float*)d_in[0];
    const float* D     = (const float*)d_in[1];
    const float* Lre   = (const float*)d_in[2];
    const float* Lim   = (const float*)d_in[3];
    const float* logdt = (const float*)d_in[4];
    const float* Wre   = (const float*)d_in[5];
    const float* Wim   = (const float*)d_in[6];
    float* out = (float*)d_out;

    char* ws = (char*)d_ws;
    _Float16* kbuf = (_Float16*)ws;                       // H*512 halfs = 1 MiB
    _Float16* uT   = (_Float16*)(ws + (1 << 20));         // B*H*ROW halfs ~= 38 MB

    const size_t ndw = (size_t)B_ * H_ * ROW / 2;         // u_t size in dwords
    dss_zfill<<<2048, 256, 0, stream>>>((uint32_t*)uT, ndw);
    dss_utrans<<<dim3(H_ / 32, L_ / 32, B_), dim3(32, 8), 0, stream>>>(u, uT);
    dss_kgen<<<H_, 128, 0, stream>>>(Lre, Lim, logdt, Wre, Wim, kbuf);
    dss_conv<<<dim3(H_, B_), 256, 0, stream>>>(uT, kbuf, u, D, out);
}
